// ExplicitC2VLayer_42709154792031
// MI455X (gfx1250) — compile-verified, matrix operand order checked by reading, not run on hardware
//
#include <hip/hip_runtime.h>
#include <hip/hip_bf16.h>
#include <hip/hip_fp16.h>

#define NN 50000
#define NE 320000
#define HDIM 128
#define NHEADS 4
#define EDIM 8
#define DEGDIM 16
#define EF (NE + NN)
#define KCAT 160   /* 144 padded to a multiple of 32 */

typedef __attribute__((ext_vector_type(16))) _Float16 v16h;
typedef __attribute__((ext_vector_type(8)))  _Float16 v8h;
typedef __attribute__((ext_vector_type(4)))  _Float16 v4h;
typedef __attribute__((ext_vector_type(8)))  float    v8f;
typedef __attribute__((ext_vector_type(4)))  float    v4f;

// ---------------- utility device helpers ----------------

__device__ inline float wave_sum32(float v) {
#pragma unroll
  for (int off = 16; off > 0; off >>= 1) v += __shfl_xor(v, off, 32);
  return v;
}

// monotonic encode of float for unsigned atomicMax (handles negatives)
__device__ inline unsigned enc_max(float f) {
  unsigned u = __float_as_uint(f);
  return (u & 0x80000000u) ? ~u : (u | 0x80000000u);
}
__device__ inline float dec_max(unsigned e) {
  unsigned u = (e & 0x80000000u) ? (e ^ 0x80000000u) : ~e;
  return __uint_as_float(u);
}

// ---------------- operand prep kernels ----------------

__global__ void f32_to_f16_kernel(const float* __restrict__ src,
                                  _Float16* __restrict__ dst, size_t n4) {
  size_t t = (size_t)blockIdx.x * blockDim.x + threadIdx.x;
  if (t >= n4) return;
  v4f v = *(const v4f*)(src + t * 4);
  v4h h = {(_Float16)v[0], (_Float16)v[1], (_Float16)v[2], (_Float16)v[3]};
  *(v4h*)(dst + t * 4) = h;
}

// Repack f32 row-major B[K,N] into fragment-ordered f16:
// Bp[((tn*kSteps + ks)*32 + lane)*16 + j] = B[kmap, tn*16 + lane%16]
// kmap = ks*32 + (j/8)*16 + (lane/16)*8 + j%8   (ISA 7.12.2 16-bit B layout)
__global__ void pack_b_kernel(const float* __restrict__ B, _Float16* __restrict__ Bp,
                              int Kp, int Kreal, int N) {
  int t = blockIdx.x * blockDim.x + threadIdx.x;
  if (t >= N * Kp) return;
  int j    = t & 15;
  int lane = (t >> 4) & 31;
  int rest = t >> 9;                 // tn * kSteps + ks
  int kSteps = Kp >> 5;
  int ks = rest % kSteps;
  int tn = rest / kSteps;
  int kk = (ks << 5) + ((j >> 3) << 4) + ((lane >> 4) << 3) + (j & 7);
  int col = (tn << 4) + (lane & 15);
  float v = (kk < Kreal) ? B[(size_t)kk * N + col] : 0.f;
  Bp[t] = (_Float16)v;
}

// ---------------- WMMA GEMM ----------------
// C[M,N] = act(A16[M,K] @ B + bias). One wave32 -> 16x64 tile (4 N-tiles, A reuse),
// explicit double-buffered fragment loads. Requires M%16==0, N%64==0, K%32==0.
// act: 0 none, 1 tanh.
__global__ __launch_bounds__(256) void wmma_gemm_kernel(
    const _Float16* __restrict__ A, const _Float16* __restrict__ Bp,
    const float* __restrict__ bias, float* __restrict__ Cf32,
    _Float16* __restrict__ Cf16, int M, int K, int N, int act)
{
  int wave = (blockIdx.x * blockDim.x + threadIdx.x) >> 5;
  int lane = threadIdx.x & 31;
  int tilesN4 = N >> 6;
  int tilesM  = M >> 4;
  if (wave >= tilesM * tilesN4) return;
  int tm  = wave / tilesN4;
  int tn4 = wave - tm * tilesN4;

  int row0  = tm << 4;
  int khalf = lane >> 4;
  int l16   = lane & 15;
  int arow  = row0 + l16;
  int kSteps = K >> 5;

  const _Float16* pa  = A + (size_t)arow * K + (khalf << 3);
  const _Float16* pb0 = Bp + (((size_t)(4 * tn4 + 0) * kSteps) << 9) + ((size_t)lane << 4);
  const _Float16* pb1 = Bp + (((size_t)(4 * tn4 + 1) * kSteps) << 9) + ((size_t)lane << 4);
  const _Float16* pb2 = Bp + (((size_t)(4 * tn4 + 2) * kSteps) << 9) + ((size_t)lane << 4);
  const _Float16* pb3 = Bp + (((size_t)(4 * tn4 + 3) * kSteps) << 9) + ((size_t)lane << 4);

  v8f z8 = {0.f, 0.f, 0.f, 0.f, 0.f, 0.f, 0.f, 0.f};
  v8f acc0 = z8, acc1 = z8, acc2 = z8, acc3 = z8;

  // prologue: fragments for ks = 0
  v8h a0 = *(const v8h*)pa;
  v8h a1 = *(const v8h*)(pa + 16);
  v16h b0 = *(const v16h*)pb0;
  v16h b1 = *(const v16h*)pb1;
  v16h b2 = *(const v16h*)pb2;
  v16h b3 = *(const v16h*)pb3;

  for (int ks = 0; ks < kSteps - 1; ++ks) {
    // double buffer: issue ks+1 loads before consuming ks fragments
    const _Float16* pan = pa + (ks + 1) * 32;
    size_t boff = (size_t)(ks + 1) << 9;
    v8h  na0 = *(const v8h*)pan;
    v8h  na1 = *(const v8h*)(pan + 16);
    v16h nb0 = *(const v16h*)(pb0 + boff);
    v16h nb1 = *(const v16h*)(pb1 + boff);
    v16h nb2 = *(const v16h*)(pb2 + boff);
    v16h nb3 = *(const v16h*)(pb3 + boff);
    if (ks + 2 < kSteps)
      __builtin_prefetch(pa + (ks + 2) * 32, 0, 1);   // global_prefetch_b8
    union { v16h v; v8h h[2]; } ua;
    ua.h[0] = a0; ua.h[1] = a1;
    acc0 = __builtin_amdgcn_wmma_f32_16x16x32_f16(false, ua.v, false, b0, (short)0, acc0, false, false);
    acc1 = __builtin_amdgcn_wmma_f32_16x16x32_f16(false, ua.v, false, b1, (short)0, acc1, false, false);
    acc2 = __builtin_amdgcn_wmma_f32_16x16x32_f16(false, ua.v, false, b2, (short)0, acc2, false, false);
    acc3 = __builtin_amdgcn_wmma_f32_16x16x32_f16(false, ua.v, false, b3, (short)0, acc3, false, false);
    a0 = na0; a1 = na1; b0 = nb0; b1 = nb1; b2 = nb2; b3 = nb3;
  }
  {
    union { v16h v; v8h h[2]; } ua;
    ua.h[0] = a0; ua.h[1] = a1;
    acc0 = __builtin_amdgcn_wmma_f32_16x16x32_f16(false, ua.v, false, b0, (short)0, acc0, false, false);
    acc1 = __builtin_amdgcn_wmma_f32_16x16x32_f16(false, ua.v, false, b1, (short)0, acc1, false, false);
    acc2 = __builtin_amdgcn_wmma_f32_16x16x32_f16(false, ua.v, false, b2, (short)0, acc2, false, false);
    acc3 = __builtin_amdgcn_wmma_f32_16x16x32_f16(false, ua.v, false, b3, (short)0, acc3, false, false);
  }

  int colb = (tn4 << 6) + l16;
  v8f accs[4] = {acc0, acc1, acc2, acc3};
#pragma unroll
  for (int i = 0; i < 4; ++i) {
    int col = colb + 16 * i;
    float bs = bias ? bias[col] : 0.f;
#pragma unroll
    for (int r = 0; r < 8; ++r) {
      int row = row0 + r + (khalf << 3);            // f32 C/D layout (ISA 7.12.2)
      float v = accs[i][r] + bs;
      if (act == 1) v = tanhf(v);
      size_t idx = (size_t)row * N + col;
      if (Cf32) Cf32[idx] = v;
      if (Cf16) Cf16[idx] = (_Float16)v;
    }
  }
}

// ---------------- elementwise / rowwise kernels ----------------

__global__ void zero_kernel(float* p, size_t n) {
  size_t t = (size_t)blockIdx.x * blockDim.x + threadIdx.x;
  if (t < n) p[t] = 0.f;
}

// x_t16 = (f16)(mask ? LayerNorm(ct)*g+b : x)   (one wave per row; lane owns 4 contiguous)
__global__ void ln_mask_select_kernel(const float* __restrict__ ct,
                                      const float* __restrict__ x,
                                      const unsigned char* __restrict__ mask,
                                      const float* __restrict__ g,
                                      const float* __restrict__ b,
                                      _Float16* __restrict__ xt, int n)
{
  int w = (blockIdx.x * blockDim.x + threadIdx.x) >> 5;
  int lane = threadIdx.x & 31;
  if (w >= n) return;
  int base = lane * 4;
  v4f v = *(const v4f*)(ct + (size_t)w * HDIM + base);
  float mean = wave_sum32(v[0] + v[1] + v[2] + v[3]) * (1.f / HDIM);
  float q = 0.f;
#pragma unroll
  for (int i = 0; i < 4; ++i) { float d = v[i] - mean; q += d * d; }
  q = wave_sum32(q);
  float rstd = rsqrtf(q * (1.f / HDIM) + 1e-5f);
  bool m = mask[w] != 0;
  v4f gv = *(const v4f*)(g + base);
  v4f bv = *(const v4f*)(b + base);
  v4f xv = *(const v4f*)(x + (size_t)w * HDIM + base);
  v4h o;
#pragma unroll
  for (int i = 0; i < 4; ++i) {
    float lnv = (v[i] - mean) * rstd * gv[i] + bv[i];
    o[i] = (_Float16)(m ? lnv : xv[i]);
  }
  *(v4h*)(xt + (size_t)w * HDIM + base) = o;
}

// dst16 = (f16) relu(LayerNorm(src)*g+b)
__global__ void ln_relu_kernel(const float* __restrict__ src,
                               const float* __restrict__ g,
                               const float* __restrict__ b,
                               _Float16* __restrict__ dst, int n)
{
  int w = (blockIdx.x * blockDim.x + threadIdx.x) >> 5;
  int lane = threadIdx.x & 31;
  if (w >= n) return;
  int base = lane * 4;
  v4f v = *(const v4f*)(src + (size_t)w * HDIM + base);
  float mean = wave_sum32(v[0] + v[1] + v[2] + v[3]) * (1.f / HDIM);
  float q = 0.f;
#pragma unroll
  for (int i = 0; i < 4; ++i) { float d = v[i] - mean; q += d * d; }
  q = wave_sum32(q);
  float rstd = rsqrtf(q * (1.f / HDIM) + 1e-5f);
  v4f gv = *(const v4f*)(g + base);
  v4f bv = *(const v4f*)(b + base);
  v4h o;
#pragma unroll
  for (int i = 0; i < 4; ++i) {
    float lnv = (v[i] - mean) * rstd * gv[i] + bv[i];
    o[i] = (_Float16)(lnv > 0.f ? lnv : 0.f);
  }
  *(v4h*)(dst + (size_t)w * HDIM + base) = o;
}

// out = LayerNorm( (acc/4) * sigmoid(g2) )
__global__ void final_kernel(const float* __restrict__ acc,
                             const float* __restrict__ g2,
                             const float* __restrict__ g,
                             const float* __restrict__ b,
                             float* __restrict__ out, int n)
{
  int w = (blockIdx.x * blockDim.x + threadIdx.x) >> 5;
  int lane = threadIdx.x & 31;
  if (w >= n) return;
  int base = lane * 4;
  v4f av = *(const v4f*)(acc + (size_t)w * HDIM + base);
  v4f gg = *(const v4f*)(g2  + (size_t)w * HDIM + base);
  v4f v;
#pragma unroll
  for (int i = 0; i < 4; ++i)
    v[i] = (av[i] * 0.25f) * (1.f / (1.f + expf(-gg[i])));
  float mean = wave_sum32(v[0] + v[1] + v[2] + v[3]) * (1.f / HDIM);
  float q = 0.f;
#pragma unroll
  for (int i = 0; i < 4; ++i) { float d = v[i] - mean; q += d * d; }
  q = wave_sum32(q);
  float rstd = rsqrtf(q * (1.f / HDIM) + 1e-5f);
  v4f gv = *(const v4f*)(g + base);
  v4f bv = *(const v4f*)(b + base);
  v4f o;
#pragma unroll
  for (int i = 0; i < 4; ++i) o[i] = (v[i] - mean) * rstd * gv[i] + bv[i];
  *(v4f*)(out + (size_t)w * HDIM + base) = o;
}

// ---------------- edge-side kernels ----------------

__global__ void loop_count_kernel(const int* __restrict__ ei,
                                  const float* __restrict__ edge_attr,
                                  float* __restrict__ cnt,
                                  float* __restrict__ loop_attr)
{
  int t = blockIdx.x * blockDim.x + threadIdx.x;
  if (t >= NE * EDIM) return;
  int e = t >> 3, k = t & 7;
  int dst = ei[NE + e];
  atomicAdd(loop_attr + (size_t)dst * EDIM + k, edge_attr[(size_t)e * EDIM + k]);
  if (k == 0) atomicAdd(cnt + dst, 1.f);
}

__global__ void loop_div_kernel(const float* __restrict__ cnt,
                                float* __restrict__ loop_attr)
{
  int t = blockIdx.x * blockDim.x + threadIdx.x;
  if (t >= NN * EDIM) return;
  int i = t >> 3;
  loop_attr[t] /= fmaxf(cnt[i], 1.f);
}

// one wave per (edge, head): alpha_raw = sum_d att * leakyrelu(xl[src]+xr[dst]+ea@We)
__global__ void attn_score_kernel(const int* __restrict__ ei,
                                  const float* __restrict__ edge_attr,
                                  const float* __restrict__ loop_attr,
                                  const float* __restrict__ xl,
                                  const float* __restrict__ xr,
                                  const float* __restrict__ We,
                                  const float* __restrict__ att,
                                  float* __restrict__ alpha_raw,
                                  unsigned* __restrict__ amax_enc)
{
  int w = (blockIdx.x * blockDim.x + threadIdx.x) >> 5;
  int lane = threadIdx.x & 31;
  if (w >= EF * NHEADS) return;
  int e = w >> 2, h = w & 3;
  int s, dn; const float* ea;
  if (e < NE) { s = ei[e]; dn = ei[NE + e]; ea = edge_attr + (size_t)e * EDIM; }
  else        { s = dn = e - NE;            ea = loop_attr + (size_t)s * EDIM; }
  v4f ea0 = *(const v4f*)(ea);
  v4f ea1 = *(const v4f*)(ea + 4);
  int base = lane * 4;
  int hd = h * HDIM + base;
  v4f xa = *(const v4f*)(xl + (size_t)s  * (NHEADS * HDIM) + hd);
  v4f xb = *(const v4f*)(xr + (size_t)dn * (NHEADS * HDIM) + hd);
  v4f ee = {0.f, 0.f, 0.f, 0.f};
#pragma unroll
  for (int k = 0; k < 4; ++k) {
    v4f w0 = *(const v4f*)(We + (size_t)k * (NHEADS * HDIM) + hd);
    v4f w1 = *(const v4f*)(We + (size_t)(k + 4) * (NHEADS * HDIM) + hd);
#pragma unroll
    for (int i = 0; i < 4; ++i) {
      ee[i] = fmaf(ea0[k], w0[i], ee[i]);
      ee[i] = fmaf(ea1[k], w1[i], ee[i]);
    }
  }
  v4f attv = *(const v4f*)(att + hd);
  float part = 0.f;
#pragma unroll
  for (int i = 0; i < 4; ++i) {
    float sv = xa[i] + xb[i] + ee[i];
    sv = sv > 0.f ? sv : 0.2f * sv;            // leaky_relu(0.2)
    part = fmaf(sv, attv[i], part);
  }
  part = wave_sum32(part);
  if (lane == 0) {
    alpha_raw[w] = part;
    atomicMax(amax_enc + (size_t)dn * NHEADS + h, enc_max(part));
  }
}

__global__ void softmax_denom_kernel(const int* __restrict__ ei,
                                     float* __restrict__ alpha,
                                     const unsigned* __restrict__ amax_enc,
                                     float* __restrict__ denom)
{
  int t = blockIdx.x * blockDim.x + threadIdx.x;
  if (t >= EF * NHEADS) return;
  int e = t >> 2, h = t & 3;
  int dn = (e < NE) ? ei[NE + e] : e - NE;
  float ea = expf(alpha[t] - dec_max(amax_enc[(size_t)dn * NHEADS + h]));
  alpha[t] = ea;
  atomicAdd(denom + (size_t)dn * NHEADS + h, ea);
}

// one wave per edge: acc[dst] += sum_h (alpha/denom)*xl[src,h,:]
__global__ void aggregate_kernel(const int* __restrict__ ei,
                                 const float* __restrict__ alpha,
                                 const float* __restrict__ denom,
                                 const float* __restrict__ xl,
                                 float* __restrict__ acc)
{
  int w = (blockIdx.x * blockDim.x + threadIdx.x) >> 5;
  int lane = threadIdx.x & 31;
  if (w >= EF) return;
  int e = w, s, dn;
  if (e < NE) { s = ei[e]; dn = ei[NE + e]; } else { s = dn = e - NE; }
  float wt[NHEADS];
#pragma unroll
  for (int h = 0; h < NHEADS; ++h)
    wt[h] = alpha[(size_t)e * NHEADS + h] / denom[(size_t)dn * NHEADS + h];
  int base = lane * 4;
  v4f sum = {0.f, 0.f, 0.f, 0.f};
#pragma unroll
  for (int h = 0; h < NHEADS; ++h) {
    v4f xv = *(const v4f*)(xl + (size_t)s * (NHEADS * HDIM) + h * HDIM + base);
#pragma unroll
    for (int i = 0; i < 4; ++i) sum[i] = fmaf(wt[h], xv[i], sum[i]);
  }
#pragma unroll
  for (int i = 0; i < 4; ++i)
    atomicAdd(acc + (size_t)dn * HDIM + base + i, sum[i]);
}

// cat16[N,160] = [acc/4 | deg_emb[deg] | 0-pad]
__global__ void build_cat_kernel(const float* __restrict__ acc,
                                 const int* __restrict__ deg,
                                 const float* __restrict__ deg_emb,
                                 _Float16* __restrict__ cat)
{
  int t = blockIdx.x * blockDim.x + threadIdx.x;
  if (t >= NN * KCAT) return;
  int row = t / KCAT, j = t - row * KCAT;
  float v;
  if (j < HDIM) v = acc[(size_t)row * HDIM + j] * 0.25f;
  else if (j < HDIM + DEGDIM) {
    int dg = deg[row];
    dg = dg < 0 ? 0 : (dg > 99 ? 99 : dg);
    v = deg_emb[dg * DEGDIM + (j - HDIM)];
  } else v = 0.f;
  cat[t] = (_Float16)v;
}

// ---------------- host-side orchestration ----------------

static inline int cdiv(long long a, long long b) { return (int)((a + b - 1) / b); }

extern "C" void kernel_launch(void* const* d_in, const int* in_sizes, int n_in,
                              void* d_out, int out_size, void* d_ws, size_t ws_size,
                              hipStream_t stream) {
  const float* x          = (const float*)d_in[0];
  const int*   edge_index = (const int*)d_in[1];
  const float* edge_attr  = (const float*)d_in[2];
  const int*   node_deg   = (const int*)d_in[3];
  const unsigned char* mask = (const unsigned char*)d_in[4];
  const float* Wl = (const float*)d_in[5];
  const float* Wr = (const float*)d_in[6];
  const float* We = (const float*)d_in[7];
  const float* att = (const float*)d_in[8];
  const float* ct_W1 = (const float*)d_in[9];
  const float* ct_b1 = (const float*)d_in[10];
  const float* ct_W2 = (const float*)d_in[11];
  const float* ct_b2 = (const float*)d_in[12];
  const float* ct_g  = (const float*)d_in[13];
  const float* ct_be = (const float*)d_in[14];
  const float* deg_emb = (const float*)d_in[15];
  const float* dg_W1 = (const float*)d_in[16];
  const float* dg_b1 = (const float*)d_in[17];
  const float* dg_g  = (const float*)d_in[18];
  const float* dg_be = (const float*)d_in[19];
  const float* dg_W2 = (const float*)d_in[20];
  const float* dg_b2 = (const float*)d_in[21];
  const float* ln_g  = (const float*)d_in[22];
  const float* ln_be = (const float*)d_in[23];
  float* out = (float*)d_out;

  // ---- workspace layout (float units; f16 buffers carved as half-density) ----
  float* ws = (float*)d_ws;
  size_t o = 0;
  float* acc       = ws + o; o += (size_t)NN * HDIM;
  float* cnt       = ws + o; o += (size_t)NN;
  float* loop_attr = ws + o; o += (size_t)NN * EDIM;
  float* denom     = ws + o; o += (size_t)NN * NHEADS;
  unsigned* amax   = (unsigned*)(ws + o); o += (size_t)NN * NHEADS;
  size_t zero_n = o;                               // zero all of the above
  float* ctf   = ws + o; o += (size_t)NN * HDIM;   // ct (pre-LN) scratch
  float* tmp1  = ws + o; o += (size_t)NN * HDIM;   // g1 (pre-LN) scratch
  float* g2f   = ws + o; o += (size_t)NN * HDIM;   // gate pre-sigmoid
  float* xl    = ws + o; o += (size_t)NN * NHEADS * HDIM;
  float* xr    = ws + o; o += (size_t)NN * NHEADS * HDIM;
  float* alpha = ws + o; o += (size_t)EF * NHEADS;
  _Float16* x16   = (_Float16*)(ws + o); o += (size_t)NN * HDIM / 2;
  _Float16* t116  = (_Float16*)(ws + o); o += (size_t)NN * HDIM / 2;
  _Float16* xt16  = (_Float16*)(ws + o); o += (size_t)NN * HDIM / 2;
  _Float16* h16   = (_Float16*)(ws + o); o += (size_t)NN * HDIM / 2;
  _Float16* cat16 = (_Float16*)(ws + o); o += (size_t)NN * KCAT / 2;
  _Float16* ctW1p = (_Float16*)(ws + o); o += (size_t)HDIM * HDIM / 2;
  _Float16* ctW2p = (_Float16*)(ws + o); o += (size_t)HDIM * HDIM / 2;
  _Float16* Wlp   = (_Float16*)(ws + o); o += (size_t)HDIM * NHEADS * HDIM / 2;
  _Float16* Wrp   = (_Float16*)(ws + o); o += (size_t)HDIM * NHEADS * HDIM / 2;
  _Float16* dgW1p = (_Float16*)(ws + o); o += (size_t)KCAT * HDIM / 2;
  _Float16* dgW2p = (_Float16*)(ws + o); o += (size_t)HDIM * HDIM / 2;
  (void)ws_size; (void)in_sizes; (void)n_in; (void)out_size;

  const int T = 256;

  // 0) zero accumulators
  zero_kernel<<<cdiv(zero_n, T), T, 0, stream>>>(ws, zero_n);

  // 1) operand prep: x -> f16, pack all weight matrices to fragment order
  f32_to_f16_kernel<<<cdiv((long long)NN * HDIM / 4, T), T, 0, stream>>>(x, x16, (size_t)NN * HDIM / 4);
  pack_b_kernel<<<cdiv(HDIM * HDIM, T), T, 0, stream>>>(ct_W1, ctW1p, HDIM, HDIM, HDIM);
  pack_b_kernel<<<cdiv(HDIM * HDIM, T), T, 0, stream>>>(ct_W2, ctW2p, HDIM, HDIM, HDIM);
  pack_b_kernel<<<cdiv(NHEADS * HDIM * HDIM, T), T, 0, stream>>>(Wl, Wlp, HDIM, HDIM, NHEADS * HDIM);
  pack_b_kernel<<<cdiv(NHEADS * HDIM * HDIM, T), T, 0, stream>>>(Wr, Wrp, HDIM, HDIM, NHEADS * HDIM);
  pack_b_kernel<<<cdiv(HDIM * KCAT, T), T, 0, stream>>>(dg_W1, dgW1p, KCAT, HDIM + DEGDIM, HDIM);
  pack_b_kernel<<<cdiv(HDIM * HDIM, T), T, 0, stream>>>(dg_W2, dgW2p, HDIM, HDIM, HDIM);

  // 2) check-node transform: t116 = f16(tanh(x@W1+b1)); ctf = t116@W2+b2
  {
    int waves = (NN / 16) * (HDIM / 64);
    wmma_gemm_kernel<<<cdiv(waves, 8), T, 0, stream>>>(x16, ctW1p, ct_b1, nullptr, t116, NN, HDIM, HDIM, 1);
    wmma_gemm_kernel<<<cdiv(waves, 8), T, 0, stream>>>(t116, ctW2p, ct_b2, ctf, nullptr, NN, HDIM, HDIM, 0);
  }
  // 3) x_t16 = mask ? LN(ct) : x
  ln_mask_select_kernel<<<cdiv(NN, 8), T, 0, stream>>>(ctf, x, mask, ct_g, ct_be, xt16, NN);

  // 4) xl = x_t@Wl, xr = x_t@Wr   (N x 512, fp32 out for edge gathers)
  {
    int waves = (NN / 16) * ((NHEADS * HDIM) / 64);
    wmma_gemm_kernel<<<cdiv(waves, 8), T, 0, stream>>>(xt16, Wlp, nullptr, xl, nullptr, NN, HDIM, NHEADS * HDIM, 0);
    wmma_gemm_kernel<<<cdiv(waves, 8), T, 0, stream>>>(xt16, Wrp, nullptr, xr, nullptr, NN, HDIM, NHEADS * HDIM, 0);
  }

  // 5) self-loop edge_attr = segment-mean over dst
  loop_count_kernel<<<cdiv((long long)NE * EDIM, T), T, 0, stream>>>(edge_index, edge_attr, cnt, loop_attr);
  loop_div_kernel<<<cdiv((long long)NN * EDIM, T), T, 0, stream>>>(cnt, loop_attr);

  // 6) attention: scores + segmax, exp + denom, weighted scatter
  attn_score_kernel<<<cdiv((long long)EF * NHEADS, 8), T, 0, stream>>>(
      edge_index, edge_attr, loop_attr, xl, xr, We, att, alpha, amax);
  softmax_denom_kernel<<<cdiv((long long)EF * NHEADS, T), T, 0, stream>>>(edge_index, alpha, amax, denom);
  aggregate_kernel<<<cdiv((long long)EF, 8), T, 0, stream>>>(edge_index, alpha, denom, xl, acc);

  // 7) degree gate
  build_cat_kernel<<<cdiv((long long)NN * KCAT, T), T, 0, stream>>>(acc, node_deg, deg_emb, cat16);
  {
    int waves = (NN / 16) * (HDIM / 64);
    wmma_gemm_kernel<<<cdiv(waves, 8), T, 0, stream>>>(cat16, dgW1p, dg_b1, tmp1, nullptr, NN, KCAT, HDIM, 0);
    ln_relu_kernel<<<cdiv(NN, 8), T, 0, stream>>>(tmp1, dg_g, dg_be, h16, NN);
    wmma_gemm_kernel<<<cdiv(waves, 8), T, 0, stream>>>(h16, dgW2p, dg_b2, g2f, nullptr, NN, HDIM, HDIM, 0);
  }

  // 8) out = LN( (acc/4) * sigmoid(g2) )
  final_kernel<<<cdiv(NN, 8), T, 0, stream>>>(acc, g2f, ln_g, ln_be, out, NN);
}
